// Attention2_2113123909987
// MI455X (gfx1250) — matrix operand order repaired and checked by reading, not compiled, not run on hardware
//
#include <hip/hip_runtime.h>
#include <hip/hip_bf16.h>

typedef __attribute__((ext_vector_type(16))) __bf16 v16bf;
typedef __attribute__((ext_vector_type(8)))  __bf16 v8bf;
typedef __attribute__((ext_vector_type(8)))  float  v8f;

union FragU { v16bf v; v8bf h[2]; };

// A-fragment (16x32 bf16): lane<16 holds row=idx, K 0..15;
// lane>=16 holds K 16..31. base must already include half*16.
__device__ __forceinline__ v16bf fragA_ld(const __bf16* base) {
    FragU f;
    f.h[0] = *(const v8bf*)(base);
    f.h[1] = *(const v8bf*)(base + 8);
    return f.v;
}
// B-fragment (32x16 bf16): lane<16 holds col=idx, K 0..15; lane>=16 K 16..31.
// base must already include half*16.
__device__ __forceinline__ v16bf fragB_ld(const __bf16* base) {
    FragU f;
    f.h[0] = *(const v8bf*)(base);
    f.h[1] = *(const v8bf*)(base + 8);
    return f.v;
}

__device__ __forceinline__ v8f wmma_bf16(v16bf a, v16bf b, v8f c) {
    return __builtin_amdgcn_wmma_f32_16x16x32_bf16(
        /*neg_a=*/false, a, /*neg_b=*/false, b,
        /*c_mod=*/(short)0, c, /*reuse_a=*/false, /*reuse_b=*/false);
}

// Async 16B copy global -> LDS (per lane). Tracked by ASYNCcnt.
// Generic pointers to LDS carry the DS offset in their low 32 bits.
__device__ __forceinline__ void async_ld_b128(__bf16* lds, const __bf16* g) {
    unsigned ldsoff = (unsigned)(size_t)lds;
    asm volatile("global_load_async_to_lds_b128 %0, %1, off"
                 :: "v"(ldsoff), "v"(g) : "memory");
}
__device__ __forceinline__ void wait_async0() {
    asm volatile("s_wait_asynccnt 0x0" ::: "memory");
}

// ---------------------------------------------------------------------------
// Conversion kernels
// ---------------------------------------------------------------------------
__global__ void cvt_bf16_kernel(const float* __restrict__ in,
                                __bf16* __restrict__ out, int n) {
    int i = blockIdx.x * blockDim.x + threadIdx.x;
    if (i < n) out[i] = (__bf16)in[i];
}

// in [rows][cols] f32 -> out [cols][rows] bf16
__global__ void cvt_transpose_kernel(const float* __restrict__ in,
                                     __bf16* __restrict__ out,
                                     int rows, int cols) {
    int i = blockIdx.x * blockDim.x + threadIdx.x;
    if (i < rows * cols) {
        int r = i / cols, c = i - r * cols;
        out[c * rows + r] = (__bf16)in[i];
    }
}

// ---------------------------------------------------------------------------
// QKV GEMM: A = xb [4096][1024] bf16, Bt = wqkvT [3072][1024] bf16
// out scattered into qkv[3][8][4096][128] bf16, Q scaled by 1/sqrt(128)
// grid (24, 32), block 256 (8 wave32). Double-buffered async LDS staging.
// B fragments preloaded so ds_load latency overlaps the WMMA stream.
// ---------------------------------------------------------------------------
__global__ __launch_bounds__(256) void qkv_gemm_kernel(
        const __bf16* __restrict__ A, const __bf16* __restrict__ Bt,
        __bf16* __restrict__ qkv) {
    __shared__ __attribute__((aligned(16))) __bf16 As[2][128 * 32];
    __shared__ __attribute__((aligned(16))) __bf16 Bs[2][128 * 32];
    const int K = 1024;
    int bm = blockIdx.y, bn = blockIdx.x;
    int t = threadIdx.x;
    int w = t >> 5, lane = t & 31, idx = lane & 15, half = lane >> 4;
    int lr = t >> 1, ls = (t & 1) * 16;

    v8f acc[8] = {};
    const __bf16* arow = A + (size_t)(bm * 128 + lr) * K + ls;
    const __bf16* brow = Bt + (size_t)(bn * 128 + lr) * K + ls;

    // prologue: stage tile 0
    async_ld_b128(As[0] + lr * 32 + ls,     arow);
    async_ld_b128(As[0] + lr * 32 + ls + 8, arow + 8);
    async_ld_b128(Bs[0] + lr * 32 + ls,     brow);
    async_ld_b128(Bs[0] + lr * 32 + ls + 8, brow + 8);
    wait_async0();
    __syncthreads();

    int cur = 0;
    for (int kk = 0; kk < K; kk += 32, cur ^= 1) {
        if (kk + 32 < K) {  // stage next tile while computing this one
            async_ld_b128(As[cur ^ 1] + lr * 32 + ls,     arow + kk + 32);
            async_ld_b128(As[cur ^ 1] + lr * 32 + ls + 8, arow + kk + 40);
            async_ld_b128(Bs[cur ^ 1] + lr * 32 + ls,     brow + kk + 32);
            async_ld_b128(Bs[cur ^ 1] + lr * 32 + ls + 8, brow + kk + 40);
        }
        // preload all fragments, then run the WMMA chain back-to-back
        v16bf a = fragA_ld(As[cur] + (w * 16 + idx) * 32 + half * 16);
        v16bf b[8];
#pragma unroll
        for (int n = 0; n < 8; n++)
            b[n] = fragB_ld(Bs[cur] + (n * 16 + idx) * 32 + half * 16);
#pragma unroll
        for (int n = 0; n < 8; n++)
            acc[n] = wmma_bf16(a, b[n], acc[n]);
        wait_async0();
        __syncthreads();
    }

    const float scale = 0.08838834764831845f;  // 1/sqrt(128)
#pragma unroll
    for (int n = 0; n < 8; n++) {
        int col = bn * 128 + n * 16 + idx;       // [0, 3072)
        int which = col >> 10;
        int rem = col & 1023;
        int head = rem >> 7;
        int dd = rem & 127;
        float sc = (which == 0) ? scale : 1.0f;
        size_t base = ((size_t)(which * 8 + head) * 4096) * 128 + dd;
#pragma unroll
        for (int r = 0; r < 8; r++) {
            int row = bm * 128 + w * 16 + r + half * 8;
            qkv[base + (size_t)row * 128] = (__bf16)(acc[n][r] * sc);
        }
    }
}

// ---------------------------------------------------------------------------
// Flash attention: one block per (query-block of 128, head). block 256.
// Q pre-scaled. qkv[3][8][4096][128] bf16 -> attn_out[4096][1024] bf16
// ---------------------------------------------------------------------------
__global__ __launch_bounds__(256) void flash_attn_kernel(
        const __bf16* __restrict__ qkv, __bf16* __restrict__ attn_out) {
    const int C = 4096, D = 128;
    __shared__ __attribute__((aligned(16))) __bf16 KPs[128 * 128];  // K tile, later P
    __shared__ __attribute__((aligned(16))) __bf16 Vts[128 * 128];  // V tile transposed [d][k]

    int qb = blockIdx.x, head = blockIdx.y;
    const __bf16* Q = qkv + (size_t)(0 * 8 + head) * C * D;
    const __bf16* Kp = qkv + (size_t)(1 * 8 + head) * C * D;
    const __bf16* Vp = qkv + (size_t)(2 * 8 + head) * C * D;

    int t = threadIdx.x;
    int w = t >> 5, lane = t & 31, idx = lane & 15, half = lane >> 4;
    int lr = t >> 1, lc0 = (t & 1) * 64;

    // Q fragments resident in registers: rows w*16..w*16+15 of this q block
    v16bf qf[4];
    int qrow = qb * 128 + w * 16 + idx;
#pragma unroll
    for (int kc = 0; kc < 4; kc++)
        qf[kc] = fragA_ld(Q + (size_t)qrow * D + kc * 32 + half * 16);

    v8f o[8] = {};
    float mloc[8], lacc[8];
#pragma unroll
    for (int r = 0; r < 8; r++) { mloc[r] = -3.0e38f; lacc[r] = 0.0f; }

    for (int kb = 0; kb < 32; kb++) {
        // K tile: async copy straight into LDS
        const __bf16* krow = Kp + (size_t)(kb * 128 + lr) * D + lc0;
#pragma unroll
        for (int j = 0; j < 64; j += 8)
            async_ld_b128(KPs + lr * 128 + lc0 + j, krow + j);
        // V tile: load to registers, scatter transposed into LDS
        const __bf16* vrow = Vp + (size_t)(kb * 128 + lr) * D + lc0;
#pragma unroll
        for (int j0 = 0; j0 < 64; j0 += 8) {
            v8bf vv = *(const v8bf*)(vrow + j0);
#pragma unroll
            for (int e = 0; e < 8; e++)
                Vts[(lc0 + j0 + e) * 128 + lr] = vv[e];
        }
        wait_async0();
        __syncthreads();

        // S = Q * K^T  (16 q rows x 128 keys per wave)
        v8f s[8] = {};
#pragma unroll
        for (int kc = 0; kc < 4; kc++) {
            v16bf a = qf[kc];
#pragma unroll
            for (int n = 0; n < 8; n++) {
                v16bf b = fragB_ld(KPs + (n * 16 + idx) * 128 + kc * 32 + half * 16);
                s[n] = wmma_bf16(a, b, s[n]);
            }
        }
        __syncthreads();  // everyone done reading K tile (P will overwrite it)

        // online softmax (rows r+half*8 live replicated across the 16-lane group)
#pragma unroll
        for (int r = 0; r < 8; r++) {
            float mx = s[0][r];
#pragma unroll
            for (int n = 1; n < 8; n++) mx = fmaxf(mx, s[n][r]);
#pragma unroll
            for (int msk = 1; msk < 16; msk <<= 1)
                mx = fmaxf(mx, __shfl_xor(mx, msk, 32));
            float mi = fmaxf(mloc[r], mx);
            float corr = __expf(mloc[r] - mi);
            mloc[r] = mi;
            float rs = 0.0f;
#pragma unroll
            for (int n = 0; n < 8; n++) {
                float p = __expf(s[n][r] - mi);
                s[n][r] = p;
                rs += p;
            }
#pragma unroll
            for (int msk = 1; msk < 16; msk <<= 1)
                rs += __shfl_xor(rs, msk, 32);
            lacc[r] = lacc[r] * corr + rs;
#pragma unroll
            for (int n = 0; n < 8; n++) o[n][r] *= corr;
        }

        // write P (bf16) into per-wave region of the (former) K tile
        __bf16* P = KPs + w * 2048;
#pragma unroll
        for (int n = 0; n < 8; n++)
#pragma unroll
            for (int r = 0; r < 8; r++)
                P[(r + half * 8) * 128 + n * 16 + idx] = (__bf16)s[n][r];
        __syncthreads();

        // O += P * V
#pragma unroll
        for (int kc = 0; kc < 4; kc++) {
            v16bf a = fragA_ld(P + idx * 128 + kc * 32 + half * 16);
#pragma unroll
            for (int n = 0; n < 8; n++) {
                v16bf b = fragB_ld(Vts + (n * 16 + idx) * 128 + kc * 32 + half * 16);
                o[n] = wmma_bf16(a, b, o[n]);
            }
        }
        __syncthreads();  // done with Vts / P before next tile load
    }

    // epilogue: O /= l, store bf16 at [row][head*128 + col]
#pragma unroll
    for (int n = 0; n < 8; n++) {
        int col = head * 128 + n * 16 + idx;
#pragma unroll
        for (int r = 0; r < 8; r++) {
            int row = qb * 128 + w * 16 + r + half * 8;
            attn_out[(size_t)row * 1024 + col] = (__bf16)(o[n][r] / lacc[r]);
        }
    }
}

// ---------------------------------------------------------------------------
// Projection GEMM: A = attn [4096][1024] bf16, Bt = wprojT [1024][1024] bf16
// out f32 [4096][1024] + bias. grid (8, 32), block 256. Double-buffered async.
// ---------------------------------------------------------------------------
__global__ __launch_bounds__(256) void proj_gemm_kernel(
        const __bf16* __restrict__ A, const __bf16* __restrict__ Bt,
        const float* __restrict__ bias, float* __restrict__ out) {
    __shared__ __attribute__((aligned(16))) __bf16 As[2][128 * 32];
    __shared__ __attribute__((aligned(16))) __bf16 Bs[2][128 * 32];
    const int K = 1024;
    int bm = blockIdx.y, bn = blockIdx.x;
    int t = threadIdx.x;
    int w = t >> 5, lane = t & 31, idx = lane & 15, half = lane >> 4;
    int lr = t >> 1, ls = (t & 1) * 16;

    v8f acc[8] = {};
    const __bf16* arow = A + (size_t)(bm * 128 + lr) * K + ls;
    const __bf16* brow = Bt + (size_t)(bn * 128 + lr) * K + ls;

    async_ld_b128(As[0] + lr * 32 + ls,     arow);
    async_ld_b128(As[0] + lr * 32 + ls + 8, arow + 8);
    async_ld_b128(Bs[0] + lr * 32 + ls,     brow);
    async_ld_b128(Bs[0] + lr * 32 + ls + 8, brow + 8);
    wait_async0();
    __syncthreads();

    int cur = 0;
    for (int kk = 0; kk < K; kk += 32, cur ^= 1) {
        if (kk + 32 < K) {
            async_ld_b128(As[cur ^ 1] + lr * 32 + ls,     arow + kk + 32);
            async_ld_b128(As[cur ^ 1] + lr * 32 + ls + 8, arow + kk + 40);
            async_ld_b128(Bs[cur ^ 1] + lr * 32 + ls,     brow + kk + 32);
            async_ld_b128(Bs[cur ^ 1] + lr * 32 + ls + 8, brow + kk + 40);
        }
        v16bf a = fragA_ld(As[cur] + (w * 16 + idx) * 32 + half * 16);
        v16bf b[8];
#pragma unroll
        for (int n = 0; n < 8; n++)
            b[n] = fragB_ld(Bs[cur] + (n * 16 + idx) * 32 + half * 16);
#pragma unroll
        for (int n = 0; n < 8; n++)
            acc[n] = wmma_bf16(a, b[n], acc[n]);
        wait_async0();
        __syncthreads();
    }

#pragma unroll
    for (int n = 0; n < 8; n++) {
        int col = bn * 128 + n * 16 + idx;
        float bb = bias[col];
#pragma unroll
        for (int r = 0; r < 8; r++) {
            int row = bm * 128 + w * 16 + r + half * 8;
            out[(size_t)row * 1024 + col] = acc[n][r] + bb;
        }
    }
}

// ---------------------------------------------------------------------------
extern "C" void kernel_launch(void* const* d_in, const int* in_sizes, int n_in,
                              void* d_out, int out_size, void* d_ws, size_t ws_size,
                              hipStream_t stream) {
    (void)in_sizes; (void)n_in; (void)out_size; (void)ws_size;
    const float* x      = (const float*)d_in[0];   // [4096][1024]
    const float* qkv_w  = (const float*)d_in[1];   // [1024][3072]
    const float* proj_w = (const float*)d_in[2];   // [1024][1024]
    const float* proj_b = (const float*)d_in[3];   // [1024]
    float* out = (float*)d_out;                    // [4096][1024]

    // workspace layout (bf16): 8 + 6 + 2 + 24 + 8 = 48 MB
    __bf16* xb     = (__bf16*)d_ws;                      // 4096*1024
    __bf16* wqkvT  = xb     + (size_t)4096 * 1024;       // 3072*1024 (transposed)
    __bf16* wprojT = wqkvT  + (size_t)3072 * 1024;       // 1024*1024 (transposed)
    __bf16* qkvb   = wprojT + (size_t)1024 * 1024;       // 3*8*4096*128
    __bf16* attn   = qkvb   + (size_t)3 * 1024 * 4096;   // 4096*1024

    cvt_bf16_kernel<<<(4096 * 1024) / 256, 256, 0, stream>>>(x, xb, 4096 * 1024);
    cvt_transpose_kernel<<<(1024 * 3072) / 256, 256, 0, stream>>>(qkv_w, wqkvT, 1024, 3072);
    cvt_transpose_kernel<<<(1024 * 1024) / 256, 256, 0, stream>>>(proj_w, wprojT, 1024, 1024);

    qkv_gemm_kernel<<<dim3(24, 32), 256, 0, stream>>>(xb, wqkvT, qkvb);
    flash_attn_kernel<<<dim3(32, 8), 256, 0, stream>>>(qkvb, attn);
    proj_gemm_kernel<<<dim3(8, 32), 256, 0, stream>>>(attn, wprojT, proj_b, out);
}
